// LSTM_60636348284956
// MI455X (gfx1250) — matrix operand order repaired and checked by reading, not compiled, not run on hardware
//
#include <hip/hip_runtime.h>
#include <hip/hip_fp16.h>

typedef __attribute__((ext_vector_type(16))) _Float16 v16h;
typedef __attribute__((ext_vector_type(8)))  _Float16 v8h;
typedef __attribute__((ext_vector_type(4)))  _Float16 v4h;
typedef __attribute__((ext_vector_type(8)))  float    v8f;
typedef __attribute__((ext_vector_type(4)))  int      v4i;
typedef __attribute__((address_space(1)))    v4i      gv4i;   // global int4
typedef __attribute__((address_space(3)))    v4i      sv4i;   // LDS int4

#define T_SEQ   4096
#define VOCAB   32000
#define EMBD    192
#define HIDDEN  512
#define NWG_LSTM 32
#define COLS_PER_WG (HIDDEN / NWG_LSTM)   // 16 columns owned per workgroup

#if __has_builtin(__builtin_amdgcn_global_load_async_to_lds_b128) && \
    __has_builtin(__builtin_amdgcn_s_wait_asynccnt)
#define HAVE_ASYNC_LDS 1
#else
#define HAVE_ASYNC_LDS 0
#endif

// K-index permutation so an A fragment (ISA 16-bit A layout: lane half 0 holds
// K {0..7,16..23}, half 1 holds K {8..15,24..31}) becomes one contiguous 32B load:
// swap bits 3 and 4 of the K index within each 32-wide K tile.
__device__ __forceinline__ int permK(int j) {
    const int o = j & 31;
    const int os = (o & ~24) | ((o & 8) << 1) | ((o & 16) >> 1);
    return (j & ~31) | os;
}

// ---------------------------------------------------------------------------
// Stage 1: embedding lookup + input projection: x_seq[t] = emb[X[t]] @ W_in + b_in
// ---------------------------------------------------------------------------
__global__ void embed_proj(const int* __restrict__ X, const float* __restrict__ emb,
                           const float* __restrict__ W_in, const float* __restrict__ b_in,
                           float* __restrict__ x_seq) {
    const int t = blockIdx.x;
    __shared__ float e[EMBD];
    const int tok = X[t];
    for (int k = threadIdx.x; k < EMBD; k += blockDim.x)
        e[k] = emb[(size_t)tok * EMBD + k];
    __syncthreads();
    for (int j = threadIdx.x; j < HIDDEN; j += blockDim.x) {
        float acc = b_in[j];
#pragma unroll 4
        for (int k = 0; k < EMBD; ++k)
            acc = fmaf(e[k], W_in[(size_t)k * HIDDEN + j], acc);
        x_seq[(size_t)t * HIDDEN + j] = acc;
    }
}

// ---------------------------------------------------------------------------
// Stage 2: transpose + cast W_head [512][32000] f32 -> Wt [32000][512] f16 so
// WMMA B fragments (16 consecutive K per lane) are single contiguous 32B loads.
// 32x32 tile staged through LDS with async global->LDS loads (ASYNCcnt path).
// ---------------------------------------------------------------------------
__global__ void whead_transpose(const float* __restrict__ W, _Float16* __restrict__ Wt) {
    __shared__ __align__(16) float tile[32][36];   // pad keeps 16B row alignment
    const int t   = threadIdx.x;
    const int k0  = blockIdx.y * 32;
    const int n0  = blockIdx.x * 32;
    const int kr  = t >> 3;          // 0..31 : K row within tile
    const int seg = t & 7;           // 0..7  : 4-float segment
    const float* src = W + (size_t)(k0 + kr) * VOCAB + n0 + seg * 4;
#if HAVE_ASYNC_LDS
    __builtin_amdgcn_global_load_async_to_lds_b128(
        (gv4i*)(uintptr_t)src,
        (sv4i*)(uintptr_t)&tile[kr][seg * 4],
        0, 0);
    __builtin_amdgcn_s_wait_asynccnt(0);
#else
    const float4 v = *(const float4*)src;
    *(float4*)&tile[kr][seg * 4] = v;
#endif
    __syncthreads();
    const int nl  = t >> 3;          // 0..31 : N row of output
    const int pos = (t & 7) * 4;     // 0..28 : K position
    v4h o;
#pragma unroll
    for (int i = 0; i < 4; ++i)
        o[i] = (_Float16)tile[pos + i][nl];
    *(v4h*)(Wt + (size_t)(n0 + nl) * HIDDEN + k0 + pos) = o;
}

// ---------------------------------------------------------------------------
// Stage 3: zero h double-buffer + device barrier state (ws is poisoned 0xAA)
// ---------------------------------------------------------------------------
__global__ void lstm_init(float* __restrict__ hbuf, unsigned* __restrict__ bar) {
    const int i = blockIdx.x * blockDim.x + threadIdx.x;
    if (i < 2 * HIDDEN) hbuf[i] = 0.0f;
    if (i < 2) bar[i] = 0u;
}

// ---------------------------------------------------------------------------
// Stage 4: persistent sequential LSTM.
// 32 workgroups x 256 threads. WG g owns hidden columns [16g, 16g+16).
// Thread t: output o = t&63 (gate = o>>4, col = o&15), K-chunk kc = t>>6.
// One device-scope phase barrier per step; h exchanged through L2 with
// agent-scope atomics, double-buffered.
// ---------------------------------------------------------------------------
__global__ void lstm_seq(const float* __restrict__ x_seq,
                         const float* __restrict__ W_f,  const float* __restrict__ b_f,
                         const float* __restrict__ W_is, const float* __restrict__ b_is,
                         const float* __restrict__ W_it, const float* __restrict__ b_it,
                         const float* __restrict__ W_o,  const float* __restrict__ b_o,
                         float* __restrict__ hbuf, _Float16* __restrict__ hsF16,
                         unsigned* bar) {
    __shared__ float hx[2 * HIDDEN];
    __shared__ float part[256];
    __shared__ float gv[64];

    const int wg   = blockIdx.x;
    const int tid  = threadIdx.x;
    const int o    = tid & 63;
    const int kc   = tid >> 6;
    const int gate = o >> 4;           // 0=f 1=i_sig 2=i_tanh 3=o
    const int col  = o & 15;
    const int j    = wg * COLS_PER_WG + col;

    const float* Wg = (gate == 0) ? W_f : (gate == 1) ? W_is : (gate == 2) ? W_it : W_o;
    const float* bg = (gate == 0) ? b_f : (gate == 1) ? b_is : (gate == 2) ? b_it : b_o;
    const float  bj = bg[j];
    const float* wp = Wg + (size_t)(kc * 256) * HIDDEN + j;

    unsigned* cnt   = bar;
    unsigned* phase = bar + 1;
    float c = 0.0f;

    for (int s = 0; s < T_SEQ; ++s) {
        float* hin = hbuf + (s & 1) * HIDDEN;
        for (int i = tid; i < 2 * HIDDEN; i += 256) {
            float v;
            if (i < HIDDEN)
                v = __hip_atomic_load(hin + i, __ATOMIC_RELAXED, __HIP_MEMORY_SCOPE_AGENT);
            else
                v = x_seq[(size_t)s * HIDDEN + (i - HIDDEN)];
            hx[i] = v;
        }
        __syncthreads();

        float acc = 0.0f;
#pragma unroll 8
        for (int k = 0; k < 256; ++k)
            acc = fmaf(hx[kc * 256 + k], wp[(size_t)k * HIDDEN], acc);
        part[tid] = acc;
        __syncthreads();

        if (tid < 64)
            gv[tid] = part[tid] + part[tid + 64] + part[tid + 128] + part[tid + 192] + bj;
        __syncthreads();

        if (tid < 16) {
            const float gf  = 1.0f / (1.0f + __expf(-gv[tid]));
            const float gis = 1.0f / (1.0f + __expf(-gv[16 + tid]));
            const float git = tanhf(gv[32 + tid]);
            const float go  = 1.0f / (1.0f + __expf(-gv[48 + tid]));
            c = c * gf + gis * git;
            const float h = tanhf(c) * go;
            const int jj = wg * COLS_PER_WG + tid;
            __hip_atomic_store(hbuf + ((s + 1) & 1) * HIDDEN + jj, h,
                               __ATOMIC_RELAXED, __HIP_MEMORY_SCOPE_AGENT);
            hsF16[(size_t)s * HIDDEN + jj] = (_Float16)h;   // row-major (contiguous K) store
            __threadfence();
        }
        __syncthreads();

        if (tid == 0) {
            const unsigned ticket =
                __hip_atomic_fetch_add(cnt, 1u, __ATOMIC_ACQ_REL, __HIP_MEMORY_SCOPE_AGENT);
            if (ticket == NWG_LSTM - 1) {
                __hip_atomic_store(cnt, 0u, __ATOMIC_RELAXED, __HIP_MEMORY_SCOPE_AGENT);
                __hip_atomic_fetch_add(phase, 1u, __ATOMIC_ACQ_REL, __HIP_MEMORY_SCOPE_AGENT);
            } else {
                while (__hip_atomic_load(phase, __ATOMIC_ACQUIRE, __HIP_MEMORY_SCOPE_AGENT)
                       < (unsigned)(s + 1))
                    __builtin_amdgcn_s_sleep(2);
            }
        }
        __syncthreads();
    }
}

// ---------------------------------------------------------------------------
// Stage 5: head GEMM logits = hs(f16, row-major [M][K]) @ Wt(f16, [N][K]) + b.
// Block = 256 threads = 8 waves, block tile 32(M) x 256(N); wave tile 16x64
// (4 accumulators, A fragment reused 4x). Every fragment is one contiguous
// aligned 32B v16h load -> 2x global_load_b128, zero LDS, zero repack movs.
// K = 512 -> 64 v_wmma_f32_16x16x32_f16 per wave.
// ---------------------------------------------------------------------------
__global__ void head_gemm(const _Float16* __restrict__ A, const _Float16* __restrict__ Bt,
                          const float* __restrict__ bias, float* __restrict__ out) {
    const int lane = threadIdx.x & 31;
    const int w    = threadIdx.x >> 5;
    const int wm   = w >> 2;             // 0..1
    const int wn   = w & 3;              // 0..3
    const int m0   = blockIdx.y * 32 + wm * 16;
    const int nb   = blockIdx.x * 256 + wn * 64;
    const int row  = lane & 15;
    const int half = lane >> 4;

    const _Float16* ap  = A  + (size_t)(m0 + row) * HIDDEN + half * 16;
    const _Float16* bp0 = Bt + (size_t)(nb + 0 * 16 + row) * HIDDEN + half * 16;
    const _Float16* bp1 = Bt + (size_t)(nb + 1 * 16 + row) * HIDDEN + half * 16;
    const _Float16* bp2 = Bt + (size_t)(nb + 2 * 16 + row) * HIDDEN + half * 16;
    const _Float16* bp3 = Bt + (size_t)(nb + 3 * 16 + row) * HIDDEN + half * 16;

    v8f acc0 = {}, acc1 = {}, acc2 = {}, acc3 = {};
#pragma unroll 4
    for (int k0 = 0; k0 < HIDDEN; k0 += 32) {
        __builtin_prefetch(ap + k0 + 256, 0, 0);
        const v16h a  = *(const v16h*)(ap + k0);
        const v16h b0 = *(const v16h*)(bp0 + k0);
        const v16h b1 = *(const v16h*)(bp1 + k0);
        const v16h b2 = *(const v16h*)(bp2 + k0);
        const v16h b3 = *(const v16h*)(bp3 + k0);
        acc0 = __builtin_amdgcn_wmma_f32_16x16x32_f16(false, a, false, b0, (short)0, acc0, false, false);
        acc1 = __builtin_amdgcn_wmma_f32_16x16x32_f16(false, a, false, b1, (short)0, acc1, false, false);
        acc2 = __builtin_amdgcn_wmma_f32_16x16x32_f16(false, a, false, b2, (short)0, acc2, false, false);
        acc3 = __builtin_amdgcn_wmma_f32_16x16x32_f16(false, a, false, b3, (short)0, acc3, false, false);
    }

    v8f accs[4] = {acc0, acc1, acc2, acc3};
#pragma unroll
    for (int jt = 0; jt < 4; ++jt) {
        const int nc = nb + jt * 16 + row;
        const float bb = bias[nc];
#pragma unroll
        for (int r = 0; r < 8; ++r) {   // C layout: VGPR r -> M = r + 8*(lane/16)
            out[(size_t)(m0 + half * 8 + r) * VOCAB + nc] = accs[jt][r] + bb;
        }
    }
}

// ---------------------------------------------------------------------------
extern "C" void kernel_launch(void* const* d_in, const int* in_sizes, int n_in,
                              void* d_out, int out_size, void* d_ws, size_t ws_size,
                              hipStream_t stream) {
    const int*   X      = (const int*)  d_in[0];
    const float* emb    = (const float*)d_in[1];
    const float* W_in   = (const float*)d_in[2];
    const float* b_in   = (const float*)d_in[3];
    const float* W_f    = (const float*)d_in[4];
    const float* b_f    = (const float*)d_in[5];
    const float* W_is   = (const float*)d_in[6];
    const float* b_is   = (const float*)d_in[7];
    const float* W_it   = (const float*)d_in[8];
    const float* b_it   = (const float*)d_in[9];
    const float* W_o    = (const float*)d_in[10];
    const float* b_o    = (const float*)d_in[11];
    const float* W_head = (const float*)d_in[12];
    const float* b_head = (const float*)d_in[13];

    char* ws = (char*)d_ws;
    float*    x_seq = (float*)ws;                                   //  8 MiB
    _Float16* hs16  = (_Float16*)(ws + 8u * 1024 * 1024);           //  4 MiB (row-major [M][K])
    _Float16* Wt16  = (_Float16*)(ws + 12u * 1024 * 1024);          // 32,768,000 B ([N][K])
    float*    hbuf  = (float*)(ws + 12u * 1024 * 1024 + 32768000u); //  4 KiB h double buffer
    unsigned* bar   = (unsigned*)(ws + 12u * 1024 * 1024 + 32768000u + 4096u);
    float* out = (float*)d_out;

    embed_proj<<<T_SEQ, 256, 0, stream>>>(X, emb, W_in, b_in, x_seq);
    whead_transpose<<<dim3(VOCAB / 32, HIDDEN / 32), 256, 0, stream>>>(W_head, Wt16);
    lstm_init<<<8, 256, 0, stream>>>(hbuf, bar);
    lstm_seq<<<NWG_LSTM, 256, 0, stream>>>(x_seq, W_f, b_f, W_is, b_is,
                                           W_it, b_it, W_o, b_o, hbuf, hs16, bar);
    head_gemm<<<dim3(VOCAB / 256, T_SEQ / 32), 256, 0, stream>>>(hs16, Wt16, b_head, out);
}
